// NTNet_3547642986645
// MI455X (gfx1250) — matrix operand order repaired and checked by reading, not compiled, not run on hardware
//
#include <hip/hip_runtime.h>

// ---------------- problem constants ----------------
#define NN   50000      // nodes per graph
#define NE   800000     // edges per graph
#define EE2  (NE + NN)  // edges + self loops
#define FIN  1433       // input features
#define KPAD 1440       // 45 * 32, zero-padded K
#define NT   45         // K tiles of 32
#define NH1  8          // heads layer 1
#define NC1  8          // channels per head layer 1
#define HC   64         // NH1*NC1
#define NC2  7          // output classes
#define NB   2          // batch
#define NEG  0.2f
#define EPSV 1e-16f

typedef __attribute__((ext_vector_type(16))) __bf16 v16bf;
typedef __attribute__((ext_vector_type(8)))  float  v8f;

// ---------------- helpers ----------------
__device__ __forceinline__ unsigned short f2bf(float f) {
  unsigned u = __float_as_uint(f);
  u += 0x7FFFu + ((u >> 16) & 1u);           // round-to-nearest-even
  return (unsigned short)(u >> 16);
}
// pack two f32 -> packed bf16 dword via v_perm
__device__ __forceinline__ unsigned pack_bf16(float lo, float hi) {
  unsigned ul = __float_as_uint(lo); ul += 0x7FFFu + ((ul >> 16) & 1u);
  unsigned uh = __float_as_uint(hi); uh += 0x7FFFu + ((uh >> 16) & 1u);
  return __builtin_amdgcn_perm(uh, ul, 0x07060302u);  // {uh[3:2], ul[3:2]}
}
// order-preserving float <-> uint mapping for atomicMax on signed floats
__device__ __forceinline__ unsigned ordf(float f) {
  unsigned u = __float_as_uint(f);
  return (u & 0x80000000u) ? ~u : (u | 0x80000000u);
}
__device__ __forceinline__ float unordf(unsigned o) {
  return (o & 0x80000000u) ? __uint_as_float(o & 0x7FFFFFFFu)
                           : __uint_as_float(~o);
}
#define ORD_NEG_INF 0x007FFFFFu   // ordf(-inf)

__device__ __forceinline__ float lrelu(float x) { return x > 0.0f ? x : NEG * x; }

__device__ __forceinline__ void edge_pair(const int* __restrict__ ei, int b, int e,
                                          int& s, int& d) {
  if (e < NE) {
    const int* eb = ei + (size_t)b * 2 * NE;
    s = eb[e];
    d = eb[NE + e];
  } else {
    s = d = e - NE;   // self loop
  }
}

// async global->LDS copy, one dword per lane (gfx1250, ASYNCcnt-tracked)
__device__ __forceinline__ void async_g2l_b32(unsigned lds_off, const float* g) {
  asm volatile("global_load_async_to_lds_b32 %0, %1, off"
               :: "v"(lds_off), "v"(g) : "memory");
}
__device__ __forceinline__ void wait_async0() {
  asm volatile("s_wait_asynccnt 0x0" ::: "memory");
}

// ======== pre-kernel: W1 [1433][64] f32 -> transposed padded bf16 [64][1440] =
__global__ void convW1t(const float* __restrict__ W1, unsigned short* __restrict__ Wt) {
  int i = blockIdx.x * blockDim.x + threadIdx.x;
  if (i >= HC * KPAD) return;
  int n = i / KPAD, k = i - n * KPAD;
  float v = (k < FIN) ? W1[(size_t)k * HC + n] : 0.0f;   // zero-pad K tail
  Wt[i] = f2bf(v);
}

// ================= layer-1 GEMM: H = X @ W1  (bf16 WMMA, f32 accum) =========
// block = 256 threads = 8 waves; tile = 128 rows x 64 cols; K chunk = 32.
// A tile: async global->LDS (f32, double buffered), bf16-converted at fragment
// assembly. B fragments: loaded straight from global bf16 Wt (L2-resident).
__global__ __launch_bounds__(256) void gemm1_wmma(const float* __restrict__ X,
                                                  const unsigned short* __restrict__ Wt,
                                                  float* __restrict__ H) {
  __shared__ __align__(16) float Af[2 * 128 * 36];   // [buf][row][k] pad 36

  const int b       = blockIdx.y;
  const int rowBase = blockIdx.x * 128;
  const float* Xb = X + (size_t)b * NN * FIN;
  float*       Hb = H + (size_t)b * NN * HC;

  const int tid  = threadIdx.x;
  const int wave = tid >> 5;
  const int lane = tid & 31;
  const int lrow = lane & 15;
  const int ahk  = (lane >> 4) << 3;    // A: K base 0 / 8 per lane half
  const int bhk  = (lane >> 4) << 3;    // B: K base 0 / 8 per lane half

  const unsigned ldsBase = (unsigned)(size_t)(void*)&Af[0];
  const size_t maxIdx = (size_t)NN * FIN - 1;

  // issue async copy of K-tile t into buffer (t&1); branchless clamped addrs
  auto issue_tile = [&](int t) {
    const int k0 = t * 32;
    const unsigned bufOff = (unsigned)((t & 1) * (128 * 36) * 4);
#pragma unroll
    for (int i = 0; i < 16; ++i) {
      int idx = tid + i * 256;
      int r = idx >> 5, kk = idx & 31;
      int grow = rowBase + r;
      if (grow >= NN) grow = NN - 1;                // clamp row
      size_t gidx = (size_t)grow * FIN + (size_t)(k0 + kk);
      if (gidx > maxIdx) gidx = maxIdx;             // clamp flat (K tail)
      unsigned loff = ldsBase + bufOff + (unsigned)((r * 36 + kk) * 4);
      async_g2l_b32(loff, Xb + gidx);
    }
  };

  v8f acc[4] = {};                      // 4 column tiles of 16

  issue_tile(0);
  wait_async0();
  __syncthreads();

  for (int it = 0; it < NT; ++it) {
    if (it + 1 < NT) issue_tile(it + 1);   // overlap copy of t+1 with compute

    const int k0 = it * 32;
    // ---- assemble A fragment: f32 LDS -> packed bf16 (ISA 7.12.2 layout) ----
    const float* ap = &Af[(it & 1) * (128 * 36) + (wave * 16 + lrow) * 36];
    union { unsigned u[8]; v16bf v; } af;
#pragma unroll
    for (int d = 0; d < 4; ++d) {
      af.u[d]     = pack_bf16(ap[ahk + 2 * d],      ap[ahk + 2 * d + 1]);
      af.u[4 + d] = pack_bf16(ap[16 + ahk + 2 * d], ap[16 + ahk + 2 * d + 1]);
    }
    // ---- B fragments straight from global bf16 (16B-aligned uint4 pairs) ----
#pragma unroll
    for (int nt = 0; nt < 4; ++nt) {
      const unsigned short* wp = Wt + (size_t)(nt * 16 + lrow) * KPAD + k0 + bhk;
      union { uint4 q[2]; v16bf v; } bfrag;
      bfrag.q[0] = *(const uint4*)wp;
      bfrag.q[1] = *(const uint4*)(wp + 16);
      acc[nt] = __builtin_amdgcn_wmma_f32_16x16x32_bf16(
          false, af.v, false, bfrag.v, (short)0, acc[nt], false, false);
    }

    wait_async0();      // own async ops for tile it+1 complete
    __syncthreads();    // all waves done: tile it+1 visible, buf (it&1) free
  }

  // ---- epilogue: C layout -> global (no bias; bias applied post-agg) ----
  const int orow0 = rowBase + wave * 16 + ((lane >> 4) << 3);
  const int ocol  = lane & 15;
#pragma unroll
  for (int nt = 0; nt < 4; ++nt) {
#pragma unroll
    for (int r = 0; r < 8; ++r) {
      int grow = orow0 + r;
      if (grow < NN) Hb[(size_t)grow * HC + nt * 16 + ocol] = acc[nt][r];
    }
  }
}

// ============ layer-1 node prep: attention coeffs + init buffers ============
__global__ void prep1(const float* __restrict__ H,
                      const float* __restrict__ a_src, const float* __restrict__ a_dst,
                      float* __restrict__ es, float* __restrict__ ed,
                      unsigned* __restrict__ m, float* __restrict__ den,
                      float* __restrict__ agg) {
  int i = blockIdx.x * blockDim.x + threadIdx.x;
  if (i >= NB * NN) return;
  const float* h = H + (size_t)i * HC;
#pragma unroll
  for (int hh = 0; hh < NH1; ++hh) {
    float s = 0.0f, d = 0.0f;
#pragma unroll
    for (int c = 0; c < NC1; ++c) {
      float v = h[hh * NC1 + c];
      s += v * a_src[hh * NC1 + c];
      d += v * a_dst[hh * NC1 + c];
    }
    es[i * NH1 + hh]  = s;
    ed[i * NH1 + hh]  = d;
    m[i * NH1 + hh]   = ORD_NEG_INF;
    den[i * NH1 + hh] = 0.0f;
  }
#pragma unroll
  for (int c = 0; c < HC; ++c) agg[(size_t)i * HC + c] = 0.0f;
}

// ================= layer-1 edge passes =================
__global__ void edge1_max(const int* __restrict__ ei, const float* __restrict__ es,
                          const float* __restrict__ ed, unsigned* __restrict__ m) {
  int t = blockIdx.x * blockDim.x + threadIdx.x;
  if (t >= NB * EE2) return;
  int b = t / EE2, e = t - b * EE2, s, d;
  edge_pair(ei, b, e, s, d);
  int ns = b * NN + s, nd = b * NN + d;
#pragma unroll
  for (int hh = 0; hh < NH1; ++hh) {
    float el = lrelu(es[ns * NH1 + hh] + ed[nd * NH1 + hh]);
    atomicMax(&m[nd * NH1 + hh], ordf(el));
  }
}

__global__ void edge1_sum(const int* __restrict__ ei, const float* __restrict__ es,
                          const float* __restrict__ ed, const unsigned* __restrict__ m,
                          float* __restrict__ den) {
  int t = blockIdx.x * blockDim.x + threadIdx.x;
  if (t >= NB * EE2) return;
  int b = t / EE2, e = t - b * EE2, s, d;
  edge_pair(ei, b, e, s, d);
  int ns = b * NN + s, nd = b * NN + d;
#pragma unroll
  for (int hh = 0; hh < NH1; ++hh) {
    float el = lrelu(es[ns * NH1 + hh] + ed[nd * NH1 + hh]);
    float ex = __expf(el - unordf(m[nd * NH1 + hh]));
    atomicAdd(&den[nd * NH1 + hh], ex);
  }
}

__global__ void edge1_agg(const int* __restrict__ ei, const float* __restrict__ es,
                          const float* __restrict__ ed, const unsigned* __restrict__ m,
                          const float* __restrict__ den, const float* __restrict__ H,
                          float* __restrict__ agg) {
  int t = blockIdx.x * blockDim.x + threadIdx.x;
  if (t >= NB * EE2) return;
  int b = t / EE2, e = t - b * EE2, s, d;
  edge_pair(ei, b, e, s, d);
  int ns = b * NN + s, nd = b * NN + d;
#pragma unroll
  for (int hh = 0; hh < NH1; ++hh) {
    float el    = lrelu(es[ns * NH1 + hh] + ed[nd * NH1 + hh]);
    float alpha = __expf(el - unordf(m[nd * NH1 + hh])) / (den[nd * NH1 + hh] + EPSV);
#pragma unroll
    for (int c = 0; c < NC1; ++c)
      atomicAdd(&agg[(size_t)nd * HC + hh * NC1 + c],
                H[(size_t)ns * HC + hh * NC1 + c] * alpha);
  }
}

// ======== post layer-1: bias + ELU, tiny GEMM -> h2, layer-2 prep ========
__global__ void post1(const float* __restrict__ agg, const float* __restrict__ b1,
                      const float* __restrict__ W2, const float* __restrict__ a2s,
                      const float* __restrict__ a2d, float* __restrict__ h2,
                      float* __restrict__ es2, float* __restrict__ ed2,
                      unsigned* __restrict__ m2, float* __restrict__ den2,
                      float* __restrict__ agg2) {
  int i = blockIdx.x * blockDim.x + threadIdx.x;
  if (i >= NB * NN) return;
  float hv[HC];
#pragma unroll
  for (int k = 0; k < HC; ++k) {
    float v = agg[(size_t)i * HC + k] + b1[k];
    hv[k] = v > 0.0f ? v : (__expf(v) - 1.0f);   // ELU
  }
  float o[NC2];
#pragma unroll
  for (int c = 0; c < NC2; ++c) {
    float s = 0.0f;
#pragma unroll
    for (int k = 0; k < HC; ++k) s += hv[k] * W2[k * NC2 + c];
    o[c] = s;
    h2[i * 8 + c] = s;
  }
  float s2 = 0.0f, d2 = 0.0f;
#pragma unroll
  for (int c = 0; c < NC2; ++c) { s2 += o[c] * a2s[c]; d2 += o[c] * a2d[c]; }
  es2[i] = s2; ed2[i] = d2;
  m2[i] = ORD_NEG_INF; den2[i] = 0.0f;
#pragma unroll
  for (int c = 0; c < 8; ++c) agg2[i * 8 + c] = 0.0f;
}

// ================= layer-2 edge passes (1 head) =================
__global__ void edge2_max(const int* __restrict__ ei, const float* __restrict__ es,
                          const float* __restrict__ ed, unsigned* __restrict__ m) {
  int t = blockIdx.x * blockDim.x + threadIdx.x;
  if (t >= NB * EE2) return;
  int b = t / EE2, e = t - b * EE2, s, d;
  edge_pair(ei, b, e, s, d);
  float el = lrelu(es[b * NN + s] + ed[b * NN + d]);
  atomicMax(&m[b * NN + d], ordf(el));
}

__global__ void edge2_sum(const int* __restrict__ ei, const float* __restrict__ es,
                          const float* __restrict__ ed, const unsigned* __restrict__ m,
                          float* __restrict__ den) {
  int t = blockIdx.x * blockDim.x + threadIdx.x;
  if (t >= NB * EE2) return;
  int b = t / EE2, e = t - b * EE2, s, d;
  edge_pair(ei, b, e, s, d);
  int nd = b * NN + d;
  float el = lrelu(es[b * NN + s] + ed[nd]);
  atomicAdd(&den[nd], __expf(el - unordf(m[nd])));
}

__global__ void edge2_agg(const int* __restrict__ ei, const float* __restrict__ es,
                          const float* __restrict__ ed, const unsigned* __restrict__ m,
                          const float* __restrict__ den, const float* __restrict__ h2,
                          float* __restrict__ agg2) {
  int t = blockIdx.x * blockDim.x + threadIdx.x;
  if (t >= NB * EE2) return;
  int b = t / EE2, e = t - b * EE2, s, d;
  edge_pair(ei, b, e, s, d);
  int ns = b * NN + s, nd = b * NN + d;
  float el    = lrelu(es[ns] + ed[nd]);
  float alpha = __expf(el - unordf(m[nd])) / (den[nd] + EPSV);
#pragma unroll
  for (int c = 0; c < NC2; ++c)
    atomicAdd(&agg2[nd * 8 + c], h2[ns * 8 + c] * alpha);
}

// ================= final: bias + log_softmax =================
__global__ void final_k(const float* __restrict__ agg2, const float* __restrict__ b2,
                        float* __restrict__ out) {
  int i = blockIdx.x * blockDim.x + threadIdx.x;
  if (i >= NB * NN) return;
  float o[NC2];
  float mx = -3.4e38f;
#pragma unroll
  for (int c = 0; c < NC2; ++c) { o[c] = agg2[i * 8 + c] + b2[c]; mx = fmaxf(mx, o[c]); }
  float s = 0.0f;
#pragma unroll
  for (int c = 0; c < NC2; ++c) s += __expf(o[c] - mx);
  float ls = logf(s);
#pragma unroll
  for (int c = 0; c < NC2; ++c) out[i * NC2 + c] = o[c] - mx - ls;
}

// ================= launch =================
extern "C" void kernel_launch(void* const* d_in, const int* in_sizes, int n_in,
                              void* d_out, int out_size, void* d_ws, size_t ws_size,
                              hipStream_t stream) {
  const float* x     = (const float*)d_in[0];
  const int*   ei    = (const int*)d_in[1];
  const float* W1    = (const float*)d_in[2];
  const float* a1s   = (const float*)d_in[3];
  const float* a1d   = (const float*)d_in[4];
  const float* b1    = (const float*)d_in[5];
  const float* W2    = (const float*)d_in[6];
  const float* a2s   = (const float*)d_in[7];
  const float* a2d   = (const float*)d_in[8];
  const float* b2    = (const float*)d_in[9];
  float* out = (float*)d_out;

  // workspace layout (floats); tail is 16B aligned for the bf16 W1t
  const size_t BN = (size_t)NB * NN;
  float* ws = (float*)d_ws;
  float*    Hws   = ws;                          // BN*64
  float*    es1   = Hws  + BN * HC;              // BN*8
  float*    ed1   = es1  + BN * NH1;             // BN*8
  unsigned* m1    = (unsigned*)(ed1 + BN * NH1); // BN*8
  float*    den1  = (float*)m1 + BN * NH1;       // BN*8
  float*    agg1  = den1 + BN * NH1;             // BN*64
  float*    h2    = agg1 + BN * HC;              // BN*8 (7 used)
  float*    es2   = h2   + BN * 8;               // BN
  float*    ed2   = es2  + BN;                   // BN
  unsigned* m2    = (unsigned*)(ed2 + BN);       // BN
  float*    den2  = (float*)m2 + BN;             // BN
  float*    agg2  = den2 + BN;                   // BN*8
  unsigned short* Wt = (unsigned short*)(agg2 + BN * 8);  // 64*1440 bf16

  convW1t<<<(HC * KPAD + 255) / 256, 256, 0, stream>>>(W1, Wt);

  dim3 gg((NN + 127) / 128, NB);
  gemm1_wmma<<<gg, 256, 0, stream>>>(x, Wt, Hws);

  const int nodeBlocks = (int)((BN + 255) / 256);
  const int edgeBlocks = (NB * EE2 + 255) / 256;

  prep1<<<nodeBlocks, 256, 0, stream>>>(Hws, a1s, a1d, es1, ed1, m1, den1, agg1);
  edge1_max<<<edgeBlocks, 256, 0, stream>>>(ei, es1, ed1, m1);
  edge1_sum<<<edgeBlocks, 256, 0, stream>>>(ei, es1, ed1, m1, den1);
  edge1_agg<<<edgeBlocks, 256, 0, stream>>>(ei, es1, ed1, m1, den1, Hws, agg1);
  post1<<<nodeBlocks, 256, 0, stream>>>(agg1, b1, W2, a2s, a2d, h2, es2, ed2, m2, den2, agg2);
  edge2_max<<<edgeBlocks, 256, 0, stream>>>(ei, es2, ed2, m2);
  edge2_sum<<<edgeBlocks, 256, 0, stream>>>(ei, es2, ed2, m2, den2);
  edge2_agg<<<edgeBlocks, 256, 0, stream>>>(ei, es2, ed2, m2, den2, h2, agg2);
  final_k<<<nodeBlocks, 256, 0, stream>>>(agg2, b2, out);
}
